// UNetMidBlockCausal3D_61864708931721
// MI455X (gfx1250) — matrix operand order repaired and checked by reading, not compiled, not run on hardware
//
#include <hip/hip_runtime.h>

// ---------------------------------------------------------------------------
// CDNA5 (gfx1250) wave32 WMMA types
// ---------------------------------------------------------------------------
typedef __bf16 bf16_t;
typedef __attribute__((ext_vector_type(16))) __bf16 v16bf;
typedef __attribute__((ext_vector_type(8)))  float  v8f;
typedef __attribute__((ext_vector_type(4)))  float  f4;

#define C_     512
#define S_     8192          // T*H*W
#define CPG_   16            // channels per group (512/32)
#define EPS_   1e-6f
#define KCONV_ 13824         // 27 * 512
#define QSCALE_ 0.04419417382415922f   // 512^-0.5

union AF { v16bf v; f4 f[2]; };

static __device__ __forceinline__ v8f wmma_bf16(v16bf a, v16bf b, v8f c) {
  // D = A(16x32 bf16) * B(32x16 bf16) + C(16x16 f32)
  return __builtin_amdgcn_wmma_f32_16x16x32_bf16(false, a, false, b, (short)0, c,
                                                 false, false);
}

static __device__ __forceinline__ float silu_f(float x) {
  return x / (1.f + __expf(-x));
}

static __device__ __forceinline__ float block_reduce_sum(float v, float* red) {
  const int tid = threadIdx.x;
  red[tid] = v; __syncthreads();
  for (int o = 512; o > 0; o >>= 1) {
    if (tid < o) red[tid] += red[tid + o];
    __syncthreads();
  }
  float r = red[0]; __syncthreads();
  return r;
}

// ---------------------------------------------------------------------------
// GroupNorm (+SiLU) : channel-major f32 input -> spatial-major bf16 output
// ---------------------------------------------------------------------------
__global__ void __launch_bounds__(1024)
gn_cm_silu_kernel(const float* __restrict__ x, const float* __restrict__ gam,
                  const float* __restrict__ bet, bf16_t* __restrict__ out)
{
  __shared__ float red[1024];
  const int g = blockIdx.x, tid = threadIdx.x;
  const float* xg = x + (size_t)g * CPG_ * S_;
  const int n = CPG_ * S_;
  float s = 0.f, s2 = 0.f;
  for (int i = tid; i < n; i += 1024) { float v = xg[i]; s += v; s2 += v * v; }
  const float sum  = block_reduce_sum(s,  red);
  const float sum2 = block_reduce_sum(s2, red);
  const float mean = sum * (1.f / n);
  const float rstd = rsqrtf(sum2 * (1.f / n) - mean * mean + EPS_);
  for (int i = tid; i < n; i += 1024) {
    const int c = g * CPG_ + (i >> 13), sp = i & (S_ - 1);
    float v = (xg[i] - mean) * rstd * gam[c] + bet[c];
    out[(size_t)sp * C_ + c] = (bf16_t)silu_f(v);
  }
}

// GroupNorm (+optional SiLU): spatial-major f32 -> spatial-major bf16
__global__ void __launch_bounds__(1024)
gn_sm_kernel(const float* __restrict__ x, const float* __restrict__ gam,
             const float* __restrict__ bet, bf16_t* __restrict__ out, int do_silu)
{
  __shared__ float red[1024];
  const int g = blockIdx.x, tid = threadIdx.x;
  const int c0 = g * CPG_;
  const int n = CPG_ * S_;
  float s = 0.f, s2 = 0.f;
  for (int i = tid; i < n; i += 1024) {
    float v = x[(size_t)(i >> 4) * C_ + c0 + (i & 15)];
    s += v; s2 += v * v;
  }
  const float sum  = block_reduce_sum(s,  red);
  const float sum2 = block_reduce_sum(s2, red);
  const float mean = sum * (1.f / n);
  const float rstd = rsqrtf(sum2 * (1.f / n) - mean * mean + EPS_);
  for (int i = tid; i < n; i += 1024) {
    const int sp = i >> 4, c = c0 + (i & 15);
    float v = (x[(size_t)sp * C_ + c] - mean) * rstd * gam[c] + bet[c];
    if (do_silu) v = silu_f(v);
    out[(size_t)sp * C_ + c] = (bf16_t)v;
  }
}

// ---------------------------------------------------------------------------
// Conv weight transform: w[co][cin][kt][ky][kx] f32 -> wb[co][tap*512+cin] bf16
// ---------------------------------------------------------------------------
__global__ void __launch_bounds__(256)
convw_kernel(const float* __restrict__ w, bf16_t* __restrict__ wb)
{
  const int i = blockIdx.x * 256 + threadIdx.x;
  if (i >= C_ * KCONV_) return;
  const int co = i / KCONV_, r = i - co * KCONV_;
  const int tap = r >> 9, cin = r & 511;
  wb[i] = (bf16_t)w[(size_t)(co * C_ + cin) * 27 + tap];
}

__global__ void __launch_bounds__(256)
cvt_bf16_kernel(const float* __restrict__ a, bf16_t* __restrict__ b, int n)
{
  const int i = blockIdx.x * 256 + threadIdx.x;
  if (i < n) b[i] = (bf16_t)a[i];
}

// V [s][c] bf16 -> Vt [c][s] bf16
__global__ void __launch_bounds__(256)
vtrans_kernel(const bf16_t* __restrict__ v, bf16_t* __restrict__ vt)
{
  const int i = blockIdx.x * 256 + threadIdx.x;   // S_*C_
  const int sp = i >> 9, c = i & 511;
  vt[(size_t)c * S_ + sp] = v[i];
}

// ---------------------------------------------------------------------------
// CausalConv3d as implicit GEMM, M=spatial N=cout K=27*512, bf16 WMMA.
//   act: [s][c] bf16, wb: [co][tap*512+cin] bf16, out: [s][co] f32
// Wave tile 64x64 (16 accum tiles).  Workgroup = 8 waves stacked in M:
//   block covers 512 spatial x 64 cout;  grid = (16 spatial super-tiles, 8 cout
//   groups).  B slice (shared by all waves) double-buffered through LDS.
// ---------------------------------------------------------------------------
__global__ void __launch_bounds__(256)
conv3d_kernel(const bf16_t* __restrict__ act, const bf16_t* __restrict__ wb,
              const float* __restrict__ bias, float* __restrict__ out)
{
  __shared__ __align__(16) bf16_t Bls[2][64][32];    // [buf][co][k] 8KB

  const int tid = threadIdx.x;
  const int wv = tid >> 5, lane = tid & 31;
  const int half = lane >> 4, lm = lane & 15;
  const int sblk = blockIdx.x;              // 0..15  (512 rows each)
  const int cobase = blockIdx.y * 64;       // 0..7
  const int co_l = tid >> 2;                // cooperative B load: 64 rows
  const int seg  = (tid & 3) * 8;           // 4 x 8 bf16 segments

  int tt[4], yy[4], xxv[4];
  #pragma unroll
  for (int mt = 0; mt < 4; ++mt) {
    const int s = sblk * 512 + wv * 64 + mt * 16 + lm;    // A-frag row (M=lm)
    tt[mt] = s >> 10; yy[mt] = (s >> 5) & 31; xxv[mt] = s & 31;
  }

  v8f acc[4][4];
  #pragma unroll
  for (int mt = 0; mt < 4; ++mt)
    #pragma unroll
    for (int nt = 0; nt < 4; ++nt)
      #pragma unroll
      for (int e = 0; e < 8; ++e) acc[mt][nt][e] = 0.f;

  // prologue: stage B for kk = 0 into buffer 0
  {
    f4 breg = *(const f4*)(wb + (size_t)(cobase + co_l) * KCONV_ + seg);
    *(f4*)(&Bls[0][co_l][seg]) = breg;
  }
  int buf = 0;

  for (int kk = 0; kk < 432; ++kk) {        // K = 432 * 32
    __syncthreads();                        // Bls[buf] visible to all waves
    if (kk + 1 < 432) {                     // stage next B into other buffer
      const int tap1 = (kk + 1) >> 4, cl1 = ((kk + 1) & 15) << 5;
      f4 breg = *(const f4*)(wb + (size_t)(cobase + co_l) * KCONV_ +
                             (tap1 << 9) + cl1 + seg);
      *(f4*)(&Bls[buf ^ 1][co_l][seg]) = breg;
    }
    const int tap = kk >> 4, cl = (kk & 15) << 5;
    const int kt = tap / 9, r9 = tap - kt * 9, ky = r9 / 3, kx = r9 - ky * 3;

    // issue all A (global) and B (LDS) fragment loads up front so the
    // scheduler can overlap latency against the 16 wmmas
    AF a[4];
    #pragma unroll
    for (int mt = 0; mt < 4; ++mt) {
      int ti = tt[mt] + kt - 2; ti = ti < 0 ? 0 : ti;                 // causal pad (2,0)
      int yi = yy[mt] + ky - 1; yi = yi < 0 ? 0 : (yi > 31 ? 31 : yi); // edge pad (1,1)
      int xi = xxv[mt] + kx - 1; xi = xi < 0 ? 0 : (xi > 31 ? 31 : xi);
      const bf16_t* ap = act + (size_t)((ti << 10) + (yi << 5) + xi) * C_ + cl;
      a[mt].f[0] = *(const f4*)(ap + half * 16);
      a[mt].f[1] = *(const f4*)(ap + half * 16 + 8);
      if ((kk & 15) != 15) __builtin_prefetch(ap + 32, 0, 3);  // next cin chunk
    }
    AF b[4];
    #pragma unroll
    for (int nt = 0; nt < 4; ++nt) {
      const bf16_t* bp = &Bls[buf][nt * 16 + lm][half * 16];
      b[nt].f[0] = *(const f4*)(bp);
      b[nt].f[1] = *(const f4*)(bp + 8);
    }
    #pragma unroll
    for (int nt = 0; nt < 4; ++nt)
      #pragma unroll
      for (int mt = 0; mt < 4; ++mt)
        acc[mt][nt] = wmma_bf16(a[mt].v, b[nt].v, acc[mt][nt]);
    buf ^= 1;
  }

  #pragma unroll
  for (int mt = 0; mt < 4; ++mt)
    #pragma unroll
    for (int nt = 0; nt < 4; ++nt) {
      const int co = cobase + nt * 16 + lm;
      const float bb = bias[co];
      #pragma unroll
      for (int e = 0; e < 8; ++e) {
        const int ss = sblk * 512 + wv * 64 + mt * 16 + half * 8 + e;
        out[(size_t)ss * C_ + co] = acc[mt][nt][e] + bb;
      }
    }
}

// ---------------------------------------------------------------------------
// Linear: out[s][o] = act[s][:] . w[o][:] + bias; bf16 WMMA
// out_f32=0: bf16 output (scaled);  out_f32=1: f32 output + residual add
// ---------------------------------------------------------------------------
__global__ void __launch_bounds__(256)
linear_kernel(const bf16_t* __restrict__ act, const bf16_t* __restrict__ w,
              const float* __restrict__ bias, void* __restrict__ outp,
              const float* __restrict__ resid, float scale, int out_f32)
{
  const int wv = threadIdx.x >> 5, lane = threadIdx.x & 31;
  const int half = lane >> 4, lm = lane & 15;
  const int stile = blockIdx.x;
  const int s = stile * 16 + lm;
  const int nb = wv * 64;

  v8f acc[4];
  #pragma unroll
  for (int j = 0; j < 4; ++j)
    #pragma unroll
    for (int e = 0; e < 8; ++e) acc[j][e] = 0.f;

  #pragma unroll 4
  for (int kk = 0; kk < 16; ++kk) {
    const int cl = kk << 5;
    const bf16_t* ap = act + (size_t)s * C_ + cl;
    AF a;
    a.f[0] = *(const f4*)(ap + half * 16);
    a.f[1] = *(const f4*)(ap + half * 16 + 8);
    AF b[4];
    #pragma unroll
    for (int j = 0; j < 4; ++j) {
      const int o = nb + j * 16 + lm;
      const bf16_t* wp = w + (size_t)o * C_ + cl + half * 16;
      b[j].f[0] = *(const f4*)(wp);
      b[j].f[1] = *(const f4*)(wp + 8);
    }
    #pragma unroll
    for (int j = 0; j < 4; ++j)
      acc[j] = wmma_bf16(a.v, b[j].v, acc[j]);
  }
  #pragma unroll
  for (int j = 0; j < 4; ++j) {
    const int o = nb + j * 16 + lm;
    const float bb = bias[o];
    #pragma unroll
    for (int e = 0; e < 8; ++e) {
      const int ss = stile * 16 + half * 8 + e;
      const float v = (acc[j][e] + bb) * scale;
      if (out_f32) ((float*)outp)[(size_t)ss * C_ + o] = v + resid[(size_t)ss * C_ + o];
      else         ((bf16_t*)outp)[(size_t)ss * C_ + o] = (bf16_t)v;
    }
  }
}

// ---------------------------------------------------------------------------
// Flash attention, single head d=512, causal per-frame (frame = s/1024).
// Q,K: [s][c] bf16 (Q prescaled), Vt: [c][s] bf16, O: [s][c] bf16.
// Block = 256 threads = 2 query groups x 4 waves (d-slice 128 per wave).
// Both groups stream the same K/V chunks in lockstep -> WGP$ reuse.
// Frame limits are multiples of 1024 -> 32-key chunks are fully allowed.
// ---------------------------------------------------------------------------
__global__ void __launch_bounds__(256)
attn_kernel(const bf16_t* __restrict__ Q, const bf16_t* __restrict__ K,
            const bf16_t* __restrict__ Vt, bf16_t* __restrict__ O)
{
  __shared__ float Spart[2][4][16][32];                 // 16 KB
  __shared__ __align__(16) bf16_t Prow[2][16][32];      // 2 KB
  __shared__ float mstate[2][16], lstate[2][16], scalebuf[2][16];

  const int tid = threadIdx.x;
  const int wv = tid >> 5, lane = tid & 31;
  const int qg = wv >> 2, wv4 = wv & 3;                 // query group, d-slice wave
  const int half = lane >> 4, lm = lane & 15;
  const int qbase = blockIdx.x * 32 + qg * 16;
  const int f = (blockIdx.x * 32) >> 10;                // frame (same for both groups)
  const int nchunks = (f + 1) << 5;                     // allowed keys = (f+1)*1024

  if (tid < 32) { mstate[tid >> 4][tid & 15] = -1e30f; lstate[tid >> 4][tid & 15] = 0.f; }
  __syncthreads();

  // preload Q A-fragments for this wave's 128-wide d-slice
  AF qf[4];
  #pragma unroll
  for (int j = 0; j < 4; ++j) {
    const bf16_t* p = Q + (size_t)(qbase + lm) * C_ + wv4 * 128 + j * 32;
    qf[j].f[0] = *(const f4*)(p + half * 16);
    qf[j].f[1] = *(const f4*)(p + half * 16 + 8);
  }

  v8f oacc[8];
  #pragma unroll
  for (int dt = 0; dt < 8; ++dt)
    #pragma unroll
    for (int e = 0; e < 8; ++e) oacc[dt][e] = 0.f;

  for (int ch = 0; ch < nchunks; ++ch) {
    const int kb = ch << 5;

    // ---- partial S = Q_slice * K_slice^T (16 q x 32 keys) ----
    // issue all 8 K fragment loads first, then the 8 wmmas
    v8f s0, s1;
    #pragma unroll
    for (int e = 0; e < 8; ++e) { s0[e] = 0.f; s1[e] = 0.f; }
    AF b0[4], b1[4];
    #pragma unroll
    for (int j = 0; j < 4; ++j) {
      const int cb = wv4 * 128 + j * 32;
      const bf16_t* p0 = K + (size_t)(kb + lm) * C_ + cb + half * 16;
      const bf16_t* p1 = K + (size_t)(kb + 16 + lm) * C_ + cb + half * 16;
      b0[j].f[0] = *(const f4*)(p0); b0[j].f[1] = *(const f4*)(p0 + 8);
      b1[j].f[0] = *(const f4*)(p1); b1[j].f[1] = *(const f4*)(p1 + 8);
    }
    #pragma unroll
    for (int j = 0; j < 4; ++j) {
      s0 = wmma_bf16(qf[j].v, b0[j].v, s0);
      s1 = wmma_bf16(qf[j].v, b1[j].v, s1);
    }
    #pragma unroll
    for (int e = 0; e < 8; ++e) {
      Spart[qg][wv4][half * 8 + e][lm]      = s0[e];
      Spart[qg][wv4][half * 8 + e][16 + lm] = s1[e];
    }
    __syncthreads();

    // ---- online softmax update (lanes 0..15 of wave 0 / wave 4) ----
    if ((tid & 127) < 16) {
      const int gq = tid >> 7;
      const int q = tid & 15;
      const float m_old = mstate[gq][q], l_old = lstate[gq][q];
      float srow[32];
      float mc = -1e30f;
      #pragma unroll
      for (int j = 0; j < 32; ++j) {
        const float v = Spart[gq][0][q][j] + Spart[gq][1][q][j] +
                        Spart[gq][2][q][j] + Spart[gq][3][q][j];
        srow[j] = v;
        mc = fmaxf(mc, v);
      }
      const float m_new = fmaxf(m_old, mc);
      const float corr  = __expf(m_old - m_new);
      float ps = 0.f;
      #pragma unroll
      for (int j = 0; j < 32; ++j) {
        const float p = __expf(srow[j] - m_new);
        Prow[gq][q][j] = (bf16_t)p;
        ps += p;
      }
      mstate[gq][q]   = m_new;
      lstate[gq][q]   = l_old * corr + ps;
      scalebuf[gq][q] = corr;
    }
    __syncthreads();

    // ---- O = diag(corr)*O + P * V_slice ----
    AF pf;
    const bf16_t* pp = &Prow[qg][lm][0];
    pf.f[0] = *(const f4*)(pp + half * 16);
    pf.f[1] = *(const f4*)(pp + half * 16 + 8);
    float sc8[8];
    #pragma unroll
    for (int e = 0; e < 8; ++e) sc8[e] = scalebuf[qg][half * 8 + e];
    #pragma unroll
    for (int dt = 0; dt < 8; ++dt) {
      const int cb = wv4 * 128 + dt * 16;
      const bf16_t* vp = Vt + (size_t)(cb + lm) * S_ + kb + half * 16;
      AF b;
      b.f[0] = *(const f4*)(vp);
      b.f[1] = *(const f4*)(vp + 8);
      #pragma unroll
      for (int e = 0; e < 8; ++e) oacc[dt][e] *= sc8[e];
      oacc[dt] = wmma_bf16(pf.v, b.v, oacc[dt]);
    }
  }

  // ---- epilogue: normalize by row sum, store bf16 [s][c] ----
  float li[8];
  #pragma unroll
  for (int e = 0; e < 8; ++e) li[e] = 1.f / lstate[qg][half * 8 + e];
  #pragma unroll
  for (int dt = 0; dt < 8; ++dt) {
    const int c = wv4 * 128 + dt * 16 + lm;
    #pragma unroll
    for (int e = 0; e < 8; ++e) {
      const int ss = qbase + half * 8 + e;
      O[(size_t)ss * C_ + c] = (bf16_t)(oacc[dt][e] * li[e]);
    }
  }
}

// ---------------------------------------------------------------------------
// residual adds
// ---------------------------------------------------------------------------
__global__ void __launch_bounds__(256)
resadd_sm_cm_kernel(const float* __restrict__ h, const float* __restrict__ xcm,
                    float* __restrict__ y)
{
  const int i = blockIdx.x * 256 + threadIdx.x;     // over S_*C_, sm index
  const int sp = i >> 9, c = i & 511;
  y[i] = h[i] + xcm[(size_t)c * S_ + sp];
}

__global__ void __launch_bounds__(256)
final_cm_kernel(const float* __restrict__ h, const float* __restrict__ y,
                float* __restrict__ out)
{
  const int i = blockIdx.x * 256 + threadIdx.x;     // over C_*S_, cm index
  const int c = i >> 13, sp = i & (S_ - 1);
  out[i] = h[(size_t)sp * C_ + c] + y[(size_t)sp * C_ + c];
}

// ---------------------------------------------------------------------------
// host launcher
// ---------------------------------------------------------------------------
extern "C" void kernel_launch(void* const* d_in, const int* in_sizes, int n_in,
                              void* d_out, int out_size, void* d_ws, size_t ws_size,
                              hipStream_t stream)
{
  (void)in_sizes; (void)n_in; (void)out_size; (void)ws_size;
  const float* X      = (const float*)d_in[0];
  const float* r0_n1s = (const float*)d_in[1];
  const float* r0_n1b = (const float*)d_in[2];
  const float* r0_w1  = (const float*)d_in[3];
  const float* r0_b1  = (const float*)d_in[4];
  const float* r0_n2s = (const float*)d_in[5];
  const float* r0_n2b = (const float*)d_in[6];
  const float* r0_w2  = (const float*)d_in[7];
  const float* r0_b2  = (const float*)d_in[8];
  const float* r1_n1s = (const float*)d_in[9];
  const float* r1_n1b = (const float*)d_in[10];
  const float* r1_w1  = (const float*)d_in[11];
  const float* r1_b1  = (const float*)d_in[12];
  const float* r1_n2s = (const float*)d_in[13];
  const float* r1_n2b = (const float*)d_in[14];
  const float* r1_w2  = (const float*)d_in[15];
  const float* r1_b2  = (const float*)d_in[16];
  const float* a_gns  = (const float*)d_in[17];
  const float* a_gnb  = (const float*)d_in[18];
  const float* a_wq   = (const float*)d_in[19];
  const float* a_bq   = (const float*)d_in[20];
  const float* a_wk   = (const float*)d_in[21];
  const float* a_bk   = (const float*)d_in[22];
  const float* a_wv   = (const float*)d_in[23];
  const float* a_bv   = (const float*)d_in[24];
  const float* a_wo   = (const float*)d_in[25];
  const float* a_bo   = (const float*)d_in[26];

  char* ws = (char*)d_ws;
  size_t off = 0;
  auto alloc = [&](size_t bytes) -> void* {
    void* p = ws + off;
    off = (off + bytes + 255) & ~(size_t)255;
    return p;
  };
  bf16_t* WB  = (bf16_t*)alloc((size_t)C_ * KCONV_ * 2);   // conv weight bf16
  bf16_t* ACT = (bf16_t*)alloc((size_t)S_ * C_ * 2);       // GN+SiLU output
  float*  HA  = (float*) alloc((size_t)S_ * C_ * 4);       // conv1 out
  float*  HB  = (float*) alloc((size_t)S_ * C_ * 4);       // conv2 out
  float*  FY  = (float*) alloc((size_t)S_ * C_ * 4);       // resnet0 out / attn resid
  float*  FY2 = (float*) alloc((size_t)S_ * C_ * 4);       // attn out / resnet1 resid
  bf16_t* QB  = (bf16_t*)alloc((size_t)S_ * C_ * 2);
  bf16_t* KB  = (bf16_t*)alloc((size_t)S_ * C_ * 2);
  bf16_t* VB  = (bf16_t*)alloc((size_t)S_ * C_ * 2);
  bf16_t* VT  = (bf16_t*)alloc((size_t)S_ * C_ * 2);
  bf16_t* OB  = (bf16_t*)alloc((size_t)S_ * C_ * 2);
  bf16_t* WL  = (bf16_t*)alloc((size_t)C_ * C_ * 2);       // linear weight bf16

  const int NWCONV = (C_ * KCONV_ + 255) / 256;            // 27648 blocks
  const int NWLIN  = (C_ * C_ + 255) / 256;                // 1024 blocks
  const int NELEM  = (S_ * C_) / 256;                      // 16384 blocks
  const dim3 CONVGRID(16, 8);                              // 512 spatial x 64 cout

  // ---------------- resnet block 0 ----------------
  gn_cm_silu_kernel<<<32, 1024, 0, stream>>>(X, r0_n1s, r0_n1b, ACT);
  convw_kernel<<<NWCONV, 256, 0, stream>>>(r0_w1, WB);
  conv3d_kernel<<<CONVGRID, 256, 0, stream>>>(ACT, WB, r0_b1, HA);
  gn_sm_kernel<<<32, 1024, 0, stream>>>(HA, r0_n2s, r0_n2b, ACT, 1);
  convw_kernel<<<NWCONV, 256, 0, stream>>>(r0_w2, WB);
  conv3d_kernel<<<CONVGRID, 256, 0, stream>>>(ACT, WB, r0_b2, HB);
  resadd_sm_cm_kernel<<<NELEM, 256, 0, stream>>>(HB, X, FY);

  // ---------------- causal frame attention ----------------
  gn_sm_kernel<<<32, 1024, 0, stream>>>(FY, a_gns, a_gnb, ACT, 0);
  cvt_bf16_kernel<<<NWLIN, 256, 0, stream>>>(a_wq, WL, C_ * C_);
  linear_kernel<<<512, 256, 0, stream>>>(ACT, WL, a_bq, (void*)QB, nullptr, QSCALE_, 0);
  cvt_bf16_kernel<<<NWLIN, 256, 0, stream>>>(a_wk, WL, C_ * C_);
  linear_kernel<<<512, 256, 0, stream>>>(ACT, WL, a_bk, (void*)KB, nullptr, 1.f, 0);
  cvt_bf16_kernel<<<NWLIN, 256, 0, stream>>>(a_wv, WL, C_ * C_);
  linear_kernel<<<512, 256, 0, stream>>>(ACT, WL, a_bv, (void*)VB, nullptr, 1.f, 0);
  vtrans_kernel<<<NELEM, 256, 0, stream>>>(VB, VT);
  attn_kernel<<<256, 256, 0, stream>>>(QB, KB, VT, OB);
  cvt_bf16_kernel<<<NWLIN, 256, 0, stream>>>(a_wo, WL, C_ * C_);
  linear_kernel<<<512, 256, 0, stream>>>(OB, WL, a_bo, (void*)FY2, FY, 1.f, 1);

  // ---------------- resnet block 1 ----------------
  gn_sm_kernel<<<32, 1024, 0, stream>>>(FY2, r1_n1s, r1_n1b, ACT, 1);
  convw_kernel<<<NWCONV, 256, 0, stream>>>(r1_w1, WB);
  conv3d_kernel<<<CONVGRID, 256, 0, stream>>>(ACT, WB, r1_b1, HA);
  gn_sm_kernel<<<32, 1024, 0, stream>>>(HA, r1_n2s, r1_n2b, ACT, 1);
  convw_kernel<<<NWCONV, 256, 0, stream>>>(r1_w2, WB);
  conv3d_kernel<<<CONVGRID, 256, 0, stream>>>(ACT, WB, r1_b2, HB);
  final_cm_kernel<<<NELEM, 256, 0, stream>>>(HB, FY2, (float*)d_out);
}
